// DeformCrossAttention2D_9208409883260
// MI455X (gfx1250) — matrix operand order repaired and checked by reading, not compiled, not run on hardware
//
#include <hip/hip_runtime.h>
#include <hip/hip_bf16.h>
#include <math.h>

// ---------------------------------------------------------------------------
// DeformCrossAttention2D for gfx1250 (MI455X), wave32 WMMA throughout.
// A fragments load as two 16B v8h runs; B fragments are gathered element-wise at stride ld.
// All WMMA inner loops are straight-line: B buffers are padded to the full
// grid coverage so no bounds checks (and no accumulator phi-copies) exist in
// any hot loop; guards live only in store epilogues. ~26 MB ws, L2-resident.
// ---------------------------------------------------------------------------

typedef _Float16 half_t;
typedef _Float16 v16h __attribute__((ext_vector_type(16)));
typedef _Float16 v8h  __attribute__((ext_vector_type(8)));
typedef float    v8f  __attribute__((ext_vector_type(8)));

#define HH     50
#define WW     50
#define HWQ    2500
#define HWB    2560            // HWQ padded to 64-tile grid coverage
#define LDX    2501            // HW + 1 (cls token at col 0)
#define DIMC   256
#define HEADS  8
#define DH     64
#define GG     4
#define INNERD 512
#define OFFD   128
#define CPB    64
#define OO     2
#define HK     12
#define JJ     144
#define JB     192             // JJ padded to 64-tile grid coverage
#define JP     160             // JJ padded to 32 for the attn@V K-loop
#define KSZ    6
#define STRD   4
#define PADC   1
#define SCL    0.125f
#define OFFSCL 4.0f

// combine two contiguous 8-half runs into a v16h fragment
static __device__ __forceinline__ v16h frag2(const half_t* p0, const half_t* p1) {
  v8h a = *(const v8h*)p0;
  v8h b = *(const v8h*)p1;
  return __builtin_shufflevector(a, b, 0,1,2,3,4,5,6,7,8,9,10,11,12,13,14,15);
}
// A fragment: lane holds row M=lane%16; halves 0..7 -> K = k0+8*hi+0..7, 8..15 -> +16
static __device__ __forceinline__ v16h ldA(const half_t* p) { return frag2(p, p + 16); }
// B fragment: lane holds N-col n0+lane%16; halves 0..7 -> K = k0+8*hi+0..7, 8..15 -> +16 (stride ld)
static __device__ __forceinline__ v16h ldB(const half_t* p, int ld) {
  v16h v;
#pragma unroll
  for (int s = 0; s < 8; ++s) { v[s] = p[(size_t)s * ld]; v[s + 8] = p[(size_t)(s + 16) * ld]; }
  return v;
}

// ---------------- small prep kernels ----------------
__global__ void cvt_f32_f16(const float* __restrict__ src, half_t* __restrict__ dst, int n) {
  int i = blockIdx.x * blockDim.x + threadIdx.x;
  if (i < n) dst[i] = (half_t)src[i];
}

// x1 (256 x 2501) -> f1h (256 x HWB) f16, cls column stripped
__global__ void strip_cls_f16(const float* __restrict__ x, half_t* __restrict__ dst) {
  int i = blockIdx.x * blockDim.x + threadIdx.x;
  if (i < DIMC * HWQ) {
    int d = i / HWQ, p = i % HWQ;
    dst[d * HWB + p] = (half_t)x[d * LDX + 1 + p];
  }
}

__global__ void copy_cls(const float* __restrict__ x1, float* __restrict__ out) {
  int o = blockIdx.x * blockDim.x + threadIdx.x;
  if (o < DIMC) out[o * LDX] = x1[o * LDX];
}

// ---------------- generic WMMA GEMM: one wave -> 16x64 C tile ----------------
// C[MxN] = A[MxK]f16 * B[KxN]f16 (+bias). Requires M%16==0, K%32==0.
// B (ldb) and Ct must be padded to gridDim.x*64 columns/rows: the k-loop is
// fully unconditional; garbage in pad columns only feeds tiles that the
// epilogue never stores (Ct pad rows are dead by construction at use sites).
__global__ __launch_bounds__(32)
void wmma_gemm(const half_t* __restrict__ A, const half_t* __restrict__ B, int ldb,
               float* __restrict__ C, int ldc, int c_off, const float* __restrict__ bias,
               half_t* __restrict__ Ch, int ldh,
               half_t* __restrict__ Ct, int ldt, float tscale,
               int M, int N, int K) {
  const int lane = threadIdx.x;
  const int mlo  = lane & 15;
  const int hi   = lane >> 4;
  const int m0   = blockIdx.y * 16;
  const int nb   = blockIdx.x * 64;
  v8f acc0 = {}, acc1 = {}, acc2 = {}, acc3 = {};
  const half_t* ap = A + (size_t)(m0 + mlo) * K + (hi << 3);
  const half_t* bp = B + (size_t)(hi << 3) * ldb + nb + mlo;
  for (int k0 = 0; k0 < K; k0 += 32) {
    v16h a = ldA(ap + k0);
    const half_t* b0 = bp + (size_t)k0 * ldb;
    v16h w0 = ldB(b0, ldb);
    v16h w1 = ldB(b0 + 16, ldb);
    v16h w2 = ldB(b0 + 32, ldb);
    v16h w3 = ldB(b0 + 48, ldb);
    acc0 = __builtin_amdgcn_wmma_f32_16x16x32_f16(false, a, false, w0, (short)0, acc0, false, false);
    acc1 = __builtin_amdgcn_wmma_f32_16x16x32_f16(false, a, false, w1, (short)0, acc1, false, false);
    acc2 = __builtin_amdgcn_wmma_f32_16x16x32_f16(false, a, false, w2, (short)0, acc2, false, false);
    acc3 = __builtin_amdgcn_wmma_f32_16x16x32_f16(false, a, false, w3, (short)0, acc3, false, false);
  }
  v8f accs[4] = {acc0, acc1, acc2, acc3};
#pragma unroll
  for (int nt = 0; nt < 4; ++nt) {
    int col = nb + nt * 16 + mlo;
    if (col < N) {                     // epilogue-only guard
#pragma unroll
      for (int r = 0; r < 8; ++r) {
        int row = m0 + r + (hi << 3);
        float v = accs[nt][r] + (bias ? bias[row] : 0.0f);
        C[(size_t)row * ldc + c_off + col] = v;
        if (Ch) Ch[(size_t)row * ldh + col] = (half_t)v;
      }
    }
    if (Ct) {                          // transposed f16 copy, unconditional
      v8h t;
#pragma unroll
      for (int r = 0; r < 8; ++r)
        t[r] = (half_t)((accs[nt][r] + (bias ? bias[m0 + r + (hi << 3)] : 0.0f)) * tscale);
      *(v8h*)(Ct + (size_t)col * ldt + m0 + (hi << 3)) = t;
    }
  }
}

// ---------------- depthwise 6x6 stride-4 conv + bias + exact GELU ----------------
__global__ void dwconv_gelu(const float* __restrict__ q, const float* __restrict__ w1,
                            const float* __restrict__ b1, float* __restrict__ off1) {
  int idx = blockIdx.x * blockDim.x + threadIdx.x;
  if (idx >= GG * OFFD * HK * HK) return;
  int x  = idx % HK;
  int y  = (idx / HK) % HK;
  int c  = (idx / (HK * HK)) % OFFD;
  int g  = idx / (HK * HK * OFFD);
  const float* ch = q + (size_t)(g * OFFD + c) * HWQ;
  const float* wk = w1 + c * KSZ * KSZ;
  float acc = 0.0f;
#pragma unroll
  for (int ky = 0; ky < KSZ; ++ky) {
    int iy = y * STRD - PADC + ky;
    if (iy < 0 || iy >= HH) continue;
#pragma unroll
    for (int kx = 0; kx < KSZ; ++kx) {
      int ix = x * STRD - PADC + kx;
      if (ix < 0 || ix >= WW) continue;
      acc = fmaf(ch[iy * WW + ix], wk[ky * KSZ + kx], acc);
    }
  }
  acc += b1[c];
  off1[idx] = 0.5f * acc * (1.0f + erff(acc * 0.70710678118654752f));
}

// ---------------- 1x1 conv (128->2) + tanh*4 + grid build + normalize ----------------
__global__ void offsets_grid(const float* __restrict__ off1, const float* __restrict__ w2,
                             float* __restrict__ vs) {
  int idx = blockIdx.x * blockDim.x + threadIdx.x;
  if (idx >= GG * JJ) return;
  int j = idx % JJ;
  int g = idx / JJ;
  int x = j % HK, y = j / HK;
  float o0 = 0.0f, o1 = 0.0f;
  const float* base = off1 + g * OFFD * JJ + j;
  for (int c = 0; c < OFFD; ++c) {
    float v = base[c * JJ];
    o0 = fmaf(v, w2[c], o0);
    o1 = fmaf(v, w2[OFFD + c], o1);
  }
  float vx = (float)x + OFFSCL * tanhf(o0);
  float vy = (float)y + OFFSCL * tanhf(o1);
  float inv = 2.0f / (float)(HK - 1);
  vs[idx * 2 + 0] = inv * vx - 1.0f;
  vs[idx * 2 + 1] = inv * vy - 1.0f;
}

// ---------------- bilinear sample of f2 (zero-pad) -> kvh f16 [256][JB] ----------------
__global__ void bilinear_sample(const float* __restrict__ x2, const float* __restrict__ vs,
                                half_t* __restrict__ kvh) {
  int idx = blockIdx.x * blockDim.x + threadIdx.x;
  if (idx >= DIMC * JJ) return;
  int j = idx % JJ;
  int d = idx / JJ;
  int g = d >> 6;
  float sx = vs[(g * JJ + j) * 2 + 0];
  float sy = vs[(g * JJ + j) * 2 + 1];
  float xf = (sx + 1.0f) * (WW * 0.5f) - 0.5f;
  float yf = (sy + 1.0f) * (HH * 0.5f) - 0.5f;
  float x0f = floorf(xf), y0f = floorf(yf);
  int x0 = (int)x0f, y0 = (int)y0f;
  float wx1 = xf - x0f, wy1 = yf - y0f;
  float wx0 = 1.0f - wx1, wy0 = 1.0f - wy1;
  const float* img = x2 + (size_t)d * LDX + 1;
  float acc = 0.0f;
#pragma unroll
  for (int dy = 0; dy < 2; ++dy)
#pragma unroll
    for (int dx = 0; dx < 2; ++dx) {
      int ix = x0 + dx, iy = y0 + dy;
      float w = (dx ? wx1 : wx0) * (dy ? wy1 : wy0);
      if (ix >= 0 && ix < WW && iy >= 0 && iy < HH)
        acc = fmaf(img[iy * WW + ix], w, acc);
    }
  kvh[d * JB + j] = (half_t)acc;
}

// ---------------- fused CPB MLP (2 -> 64 VALU, 64x64 WMMA, 64 -> 2 VALU) ----------------
__global__ __launch_bounds__(32)
void cpb_mlp(const float* __restrict__ vs,
             const float* __restrict__ w1, const float* __restrict__ b1,
             const half_t* __restrict__ w2h, const float* __restrict__ b2,
             const float* __restrict__ w3, const float* __restrict__ b3,
             float* __restrict__ biasb) {
  __shared__ float fx[16], fy[16];
  __shared__ __align__(16) half_t h1s[16 * 64];
  __shared__ float h2s[16 * 64];
  const int lane = threadIdx.x;
  const int mlo = lane & 15, hi = lane >> 4;
  const long long R = (long long)GG * HWQ * JJ;
  const long long row0 = (long long)blockIdx.x * 16;

  if (lane < 16) {
    long long row = row0 + lane;
    float px = 0.0f, py = 0.0f;
    if (row < R) {
      int j = (int)(row % JJ);
      int i = (int)((row / JJ) % HWQ);
      int g = (int)(row / ((long long)HWQ * JJ));
      float qx = 2.0f * (float)(i % WW) / (float)(WW - 1) - 1.0f;
      float qy = 2.0f * (float)(i / WW) / (float)(HH - 1) - 1.0f;
      float dx = qx - vs[(g * JJ + j) * 2 + 0];
      float dy = qy - vs[(g * JJ + j) * 2 + 1];
      px = copysignf(log1pf(fabsf(dx)), dx);
      py = copysignf(log1pf(fabsf(dy)), dy);
    }
    fx[lane] = px; fy[lane] = py;
  }
  __syncthreads();

  for (int t = lane; t < 16 * 64; t += 32) {
    int r = t >> 6, c = t & 63;
    float v = fmaf(fx[r], w1[c], fmaf(fy[r], w1[64 + c], b1[c]));
    h1s[t] = (half_t)fmaxf(v, 0.0f);
  }
  __syncthreads();

#pragma unroll
  for (int nt = 0; nt < 4; ++nt) {
    v8f acc = {};
#pragma unroll
    for (int kk = 0; kk < 2; ++kk) {
      v16h a = ldA(h1s + mlo * 64 + kk * 32 + (hi << 3));
      v16h b = ldB(w2h + (kk * 32 + (hi << 3)) * 64 + nt * 16 + mlo, 64);
      acc = __builtin_amdgcn_wmma_f32_16x16x32_f16(false, a, false, b, (short)0, acc, false, false);
    }
    int col = nt * 16 + mlo;
#pragma unroll
    for (int r = 0; r < 8; ++r)
      h2s[(r + (hi << 3)) * 64 + col] = fmaxf(acc[r] + b2[col], 0.0f);
  }
  __syncthreads();

  {
    int r = lane & 15, o = lane >> 4;
    long long row = row0 + r;
    if (row < R) {
      float acc = b3[o];
      for (int c = 0; c < 64; ++c)
        acc = fmaf(h2s[r * 64 + c], w3[c * OO + o], acc);
      int j = (int)(row % JJ);
      int i = (int)((row / JJ) % HWQ);
      int g = (int)(row / ((long long)HWQ * JJ));
      biasb[((long long)(g * OO + o) * HWQ + i) * JJ + j] = acc;
    }
  }
}

// ---------------- fused attention (sim WMMA -> +bias -> softmax -> attn@V WMMA) --------
// qT: [HWB][INNERD] f16 pre-scaled; kh: [INNERD][JJ] f16; vT: [JB][INNERD] f16
// (rows 144..159 of vT may hold garbage: attn LDS is zero for k>=144, so they
//  multiply by 0.0 and contribute nothing).
__global__ __launch_bounds__(32)
void attention(const half_t* __restrict__ qT, const half_t* __restrict__ kh,
               const half_t* __restrict__ vT, const float* __restrict__ biasb,
               half_t* __restrict__ inner_h) {
  __shared__ float sims[16 * JP];
  __shared__ __align__(16) half_t attn[16 * JP];
  const int lane = threadIdx.x;
  const int mlo = lane & 15, hi = lane >> 4;
  const int i0 = blockIdx.x * 16;
  const int h  = blockIdx.y;

  // sim = (q*scale)^T @ k : 9 N-tiles x 2 k-steps
#pragma unroll
  for (int jt = 0; jt < 9; ++jt) {
    v8f acc = {};
#pragma unroll
    for (int kk = 0; kk < 2; ++kk) {
      v16h a = ldA(qT + (size_t)(i0 + mlo) * INNERD + h * DH + kk * 32 + (hi << 3));
      v16h b = ldB(kh + (size_t)(h * DH + kk * 32 + (hi << 3)) * JJ + jt * 16 + mlo, JJ);
      acc = __builtin_amdgcn_wmma_f32_16x16x32_f16(false, a, false, b, (short)0, acc, false, false);
    }
    int col = jt * 16 + mlo;
#pragma unroll
    for (int r = 0; r < 8; ++r)
      sims[(r + (hi << 3)) * JP + col] = acc[r];
  }
  __syncthreads();

  for (int t = lane; t < 16 * JJ; t += 32) {
    int m = t / JJ, j = t % JJ;
    int i = i0 + m;
    if (i < HWQ)
      sims[m * JP + j] += biasb[((long long)h * HWQ + i) * JJ + j];
  }
  __syncthreads();

  if (lane < 16) {
    float mx = -1e30f;
    for (int j = 0; j < JJ; ++j) mx = fmaxf(mx, sims[lane * JP + j]);
    float s = 0.0f;
    for (int j = 0; j < JJ; ++j) s += __expf(sims[lane * JP + j] - mx);
    float inv = 1.0f / s;
    for (int j = 0; j < JJ; ++j)
      attn[lane * JP + j] = (half_t)(__expf(sims[lane * JP + j] - mx) * inv);
    for (int j = JJ; j < JP; ++j) attn[lane * JP + j] = (half_t)0.0f;
  }
  __syncthreads();

  // out = attn @ vT : 4 N-tiles (d) x 5 k-steps (j, zero-padded to 160)
#pragma unroll
  for (int nt = 0; nt < 4; ++nt) {
    v8f acc = {};
#pragma unroll
    for (int kk = 0; kk < 5; ++kk) {
      v16h a = ldA(attn + mlo * JP + kk * 32 + (hi << 3));
      v16h b = ldB(vT + (size_t)(kk * 32 + (hi << 3)) * INNERD + h * DH + nt * 16 + mlo, INNERD);
      acc = __builtin_amdgcn_wmma_f32_16x16x32_f16(false, a, false, b, (short)0, acc, false, false);
    }
    int d = nt * 16 + mlo;
    v8h t;
#pragma unroll
    for (int r = 0; r < 8; ++r) t[r] = (half_t)acc[r];
    *(v8h*)(inner_h + (size_t)(h * DH + d) * HWB + i0 + (hi << 3)) = t;
  }
}

// ---------------------------------------------------------------------------
static inline size_t alignup(size_t x) { return (x + 255) & ~(size_t)255; }
static inline int cdiv(int a, int b) { return (a + b - 1) / b; }

extern "C" void kernel_launch(void* const* d_in, const int* in_sizes, int n_in,
                              void* d_out, int out_size, void* d_ws, size_t ws_size,
                              hipStream_t stream) {
  const float* x1    = (const float*)d_in[0];
  const float* x2    = (const float*)d_in[1];
  const float* woff1 = (const float*)d_in[2];
  const float* boff1 = (const float*)d_in[3];
  const float* woff2 = (const float*)d_in[4];
  const float* wq    = (const float*)d_in[5];
  const float* wk    = (const float*)d_in[6];
  const float* wv    = (const float*)d_in[7];
  const float* wout  = (const float*)d_in[8];
  const float* bout  = (const float*)d_in[9];
  const float* cw1   = (const float*)d_in[10];
  const float* cb1   = (const float*)d_in[11];
  const float* cw2   = (const float*)d_in[12];
  const float* cb2   = (const float*)d_in[13];
  const float* cw3   = (const float*)d_in[14];
  const float* cb3   = (const float*)d_in[15];
  float* out = (float*)d_out;

  char* p = (char*)d_ws;
  auto take = [&](size_t bytes) { char* r = p; p += alignup(bytes); return r; };
  float*  qbuf   = (float*) take((size_t)INNERD * HWQ * 4);   // q f32 for conv branch
  half_t* qT     = (half_t*)take((size_t)HWB * INNERD * 2);   // q^T * scale, f16
  half_t* f1h    = (half_t*)take((size_t)DIMC * HWB * 2);
  half_t* wqh    = (half_t*)take((size_t)INNERD * DIMC * 2);
  half_t* wkh    = (half_t*)take((size_t)INNERD * DIMC * 2);
  half_t* wvh    = (half_t*)take((size_t)INNERD * DIMC * 2);
  half_t* wouth  = (half_t*)take((size_t)DIMC * INNERD * 2);
  half_t* w2h    = (half_t*)take((size_t)CPB * CPB * 2);
  float*  off1   = (float*) take((size_t)GG * OFFD * JJ * 4);
  float*  vs     = (float*) take((size_t)GG * JJ * 2 * 4);
  half_t* kvh    = (half_t*)take((size_t)DIMC * JB * 2);
  float*  kf     = (float*) take((size_t)INNERD * JJ * 4);
  float*  vf     = (float*) take((size_t)INNERD * JJ * 4);
  half_t* kh     = (half_t*)take((size_t)INNERD * JJ * 2);    // k f16 row-major
  half_t* vT     = (half_t*)take((size_t)JB * INNERD * 2);    // v^T f16 (padded rows)
  float*  biasb  = (float*) take((size_t)HEADS * HWQ * JJ * 4);
  half_t* innerh = (half_t*)take((size_t)INNERD * HWB * 2);
  (void)ws_size; (void)in_sizes; (void)n_in; (void)out_size;

  cvt_f32_f16<<<cdiv(INNERD * DIMC, 256), 256, 0, stream>>>(wq,  wqh, INNERD * DIMC);
  cvt_f32_f16<<<cdiv(INNERD * DIMC, 256), 256, 0, stream>>>(wk,  wkh, INNERD * DIMC);
  cvt_f32_f16<<<cdiv(INNERD * DIMC, 256), 256, 0, stream>>>(wv,  wvh, INNERD * DIMC);
  cvt_f32_f16<<<cdiv(DIMC * INNERD, 256), 256, 0, stream>>>(wout, wouth, DIMC * INNERD);
  cvt_f32_f16<<<cdiv(CPB * CPB, 256), 256, 0, stream>>>(cw2, w2h, CPB * CPB);
  strip_cls_f16<<<cdiv(DIMC * HWQ, 256), 256, 0, stream>>>(x1, f1h);

  // q = wq @ f1; also emit qT = (q*scale)^T f16 (2560 rows, unconditional store)
  wmma_gemm<<<dim3(cdiv(HWQ, 64), INNERD / 16), 32, 0, stream>>>(
      wqh, f1h, HWB, qbuf, HWQ, 0, (const float*)nullptr,
      (half_t*)nullptr, 0, qT, INNERD, SCL, INNERD, HWQ, DIMC);

  dwconv_gelu<<<cdiv(GG * OFFD * HK * HK, 256), 256, 0, stream>>>(qbuf, woff1, boff1, off1);
  offsets_grid<<<cdiv(GG * JJ, 128), 128, 0, stream>>>(off1, woff2, vs);
  bilinear_sample<<<cdiv(DIMC * JJ, 256), 256, 0, stream>>>(x2, vs, kvh);

  // k = wk @ kv (f16 row-major copy);  v = wv @ kv (f16 transposed copy)
  wmma_gemm<<<dim3(cdiv(JJ, 64), INNERD / 16), 32, 0, stream>>>(
      wkh, kvh, JB, kf, JJ, 0, (const float*)nullptr,
      kh, JJ, (half_t*)nullptr, 0, 1.0f, INNERD, JJ, DIMC);
  wmma_gemm<<<dim3(cdiv(JJ, 64), INNERD / 16), 32, 0, stream>>>(
      wvh, kvh, JB, vf, JJ, 0, (const float*)nullptr,
      (half_t*)nullptr, 0, vT, INNERD, 1.0f, INNERD, JJ, DIMC);

  // CPB relative-position-bias MLP (dominant FLOPs)
  {
    long long R = (long long)GG * HWQ * JJ;
    cpb_mlp<<<(int)((R + 15) / 16), 32, 0, stream>>>(vs, cw1, cb1, w2h, cb2, cw3, cb3, biasb);
  }

  attention<<<dim3(cdiv(HWQ, 16), HEADS), 32, 0, stream>>>(qT, kh, vT, biasb, innerh);

  // output projection + bias into d_out cols 1..2500; cls passthrough
  wmma_gemm<<<dim3(cdiv(HWQ, 64), DIMC / 16), 32, 0, stream>>>(
      wouth, innerh, HWB, out, LDX, 1, bout,
      (half_t*)nullptr, 0, (half_t*)nullptr, 0, 1.0f, DIMC, HWQ, INNERD);
  copy_cls<<<1, DIMC, 0, stream>>>(x1, out);
}
